// ONN_87359634800800
// MI455X (gfx1250) — matrix operand order repaired and checked by reading, not compiled, not run on hardware
//
#include <hip/hip_runtime.h>
#include <hip/hip_bf16.h>
#include <math.h>

typedef __attribute__((ext_vector_type(16))) _Float16 v16h;
typedef __attribute__((ext_vector_type(8)))  _Float16 v8h;
typedef __attribute__((ext_vector_type(8)))  float    v8f;

#define NUM_FIELDS  16
#define VOCAB       50000
#define TOTAL_VOCAB 800000
#define ED          32
#define BATCH       4096
#define D0          400
#define D1          400
#define IN_DIM      632     // 16*32 + 120
#define K0PAD       640     // IN_DIM padded to multiple of 32
#define K1PAD       416     // 400 padded to multiple of 32

// ---------------- workspace layout (bytes) ----------------
static const size_t OFF_H0  = 0;                                   // f16 [4096][640]
static const size_t SZ_H0   = (size_t)BATCH * K0PAD * 2;           // 5,242,880
static const size_t OFF_W0T = OFF_H0 + SZ_H0;                      // f16 [400][640]
static const size_t SZ_W0T  = (size_t)D0 * K0PAD * 2;              // 512,000
static const size_t OFF_W1T = OFF_W0T + SZ_W0T;                    // f16 [400][416]
static const size_t SZ_W1T  = (size_t)D1 * K1PAD * 2;              // 332,800
static const size_t OFF_S0  = OFF_W1T + SZ_W1T;
static const size_t OFF_T0  = OFF_S0 + D0 * 4;
static const size_t OFF_S1  = OFF_T0 + D0 * 4;
static const size_t OFF_T1  = OFF_S1 + D1 * 4;
static const size_t OFF_H1  = OFF_T1 + D1 * 4;                     // f16 [4096][416]
static const size_t SZ_H1   = (size_t)BATCH * K1PAD * 2;           // 3,407,872
static const size_t OFF_H2  = OFF_H1 + SZ_H1;                      // f32 [4096][400]
static const size_t SZ_H2   = (size_t)BATCH * D1 * 4;

// ---------------- BN fold: s = g*rsqrt(v+eps), t = (b-m)*s + bb ----------------
__global__ void onn_fold_bn(const float* __restrict__ b, const float* __restrict__ g,
                            const float* __restrict__ bb, const float* __restrict__ m,
                            const float* __restrict__ v,
                            float* __restrict__ s, float* __restrict__ t, int n)
{
    int i = blockIdx.x * blockDim.x + threadIdx.x;
    if (i < n) {
        float sc = g[i] * rsqrtf(v[i] + 1e-3f);
        s[i] = sc;
        t[i] = (b[i] - m[i]) * sc + bb[i];
    }
}

// ---------------- weight transpose + f16 convert, K padded with zeros ----------------
// src: [K][N] f32 row-major; dst: [N][Kpad] f16 row-major
__global__ void onn_wt_cvt(const float* __restrict__ w, _Float16* __restrict__ wt,
                           int K, int N, int Kpad)
{
    int idx = blockIdx.x * blockDim.x + threadIdx.x;
    if (idx >= N * Kpad) return;
    int n = idx / Kpad;
    int k = idx - n * Kpad;
    wt[idx] = (k < K) ? (_Float16)w[(size_t)k * N + n] : (_Float16)0.f;
}

// ---------------- embedding gather + FFM interactions -> h0 f16 [4096][640] ----------------
__global__ __launch_bounds__(256) void onn_gather_ffm(
    const int* __restrict__ x, const float* __restrict__ emb,
    _Float16* __restrict__ h0)
{
    const int b = blockIdx.x;
    const int t = threadIdx.x;                 // 0..255
    __shared__ float V[NUM_FIELDS][NUM_FIELDS][ED];   // [field][op][e] = 32 KB
    __shared__ int sidx[NUM_FIELDS];
    if (t < NUM_FIELDS) sidx[t] = x[b * NUM_FIELDS + t] + t * VOCAB;
    __syncthreads();

    // Stage all 256 vectors: 4 threads per vector, each thread 32 B (coalesced 128-B rows).
#pragma unroll
    for (int pass = 0; pass < 4; ++pass) {
        int vec = pass * 64 + (t >> 2);        // 0..255
        int f = vec >> 4, p = vec & 15, q = t & 3;
        const float* src = emb + ((size_t)p * TOTAL_VOCAB + (size_t)sidx[f]) * ED + q * 8;
        float4 a = *(const float4*)src;
        float4 c = *(const float4*)(src + 4);
        float* dst = &V[f][p][q * 8];
        *(float4*)dst = a;
        *(float4*)(dst + 4) = c;
    }
    __syncthreads();

    _Float16* hrow = h0 + (size_t)b * K0PAD;

    // raw: all_e[15] flattened -> cols 0..511
    int c0 = t * 2;
    hrow[c0]     = (_Float16)V[c0 >> 5][15][c0 & 31];
    hrow[c0 + 1] = (_Float16)V[(c0 + 1) >> 5][15][(c0 + 1) & 31];

    // ffm: 120 upper-triangle pairs (np.triu_indices order) -> cols 512..631; pad 632..639 = 0
    if (t < 120) {
        int i = 0, rem = t;
        while (rem >= 15 - i) { rem -= 15 - i; ++i; }
        int j = i + 1 + rem;
        const float* ea = V[i][j - 1];   // e_ij = all_e[j-1, b, i]
        const float* eb = V[j][i];       // e_ji = all_e[i,   b, j]
        float s = 0.f;
#pragma unroll
        for (int e = 0; e < ED; e += 4) {
            float4 av = *(const float4*)(ea + e);
            float4 bv = *(const float4*)(eb + e);
            s += av.x * bv.x + av.y * bv.y + av.z * bv.z + av.w * bv.w;
        }
        hrow[512 + t] = (_Float16)s;
    } else if (t < 128) {
        hrow[512 + t] = (_Float16)0.f;
    }
}

// ---------------- WMMA GEMM + folded BN + ReLU ----------------
// A: f16 [M][K] row-major. Bt: f16 [N][K] row-major (transposed weights).
// One wave computes a 32x16 output tile (two 16x16 WMMA accumulators sharing B).
union AFrag { v16h v; v8h h[2]; };

__global__ __launch_bounds__(256) void onn_gemm_bn_relu(
    const _Float16* __restrict__ A, const _Float16* __restrict__ Bt,
    const float* __restrict__ scale, const float* __restrict__ bias,
    _Float16* __restrict__ outH, float* __restrict__ outF,
    int M, int N, int K, int ldc)
{
    const int lane   = threadIdx.x & 31;
    const int waveId = blockIdx.x * (blockDim.x >> 5) + (threadIdx.x >> 5);
    const int mTiles = M >> 5;
    if (waveId >= mTiles * (N >> 4)) return;
    const int mt = waveId % mTiles;
    const int nt = waveId / mTiles;
    const int row0 = mt << 5;
    const int col0 = nt << 4;

    const int lm = lane & 15;
    const int lh = lane >> 4;
    // A fragment, contiguous k order used here (not ISA table order; B matches): lane<16 holds K 0..15; lane>=16 holds K 16..31
    const _Float16* a0p = A + (size_t)(row0 + lm) * K + lh * 16;
    const _Float16* a1p = a0p + (size_t)16 * K;
    // B fragment, same contiguous k order as A: lane = column, lane<16 K 0..15, lane>=16 K 16..31
    const _Float16* bp  = Bt + (size_t)(col0 + lm) * K + lh * 16;

    v8f acc0 = {};
    v8f acc1 = {};
    for (int k0 = 0; k0 < K; k0 += 32) {
        v16h bf = *(const v16h*)(bp + k0);
        AFrag a0;
        a0.h[0] = *(const v8h*)(a0p + k0);
        a0.h[1] = *(const v8h*)(a0p + k0 + 8);
        AFrag a1;
        a1.h[0] = *(const v8h*)(a1p + k0);
        a1.h[1] = *(const v8h*)(a1p + k0 + 8);
        acc0 = __builtin_amdgcn_wmma_f32_16x16x32_f16(false, a0.v, false, bf,
                                                      (short)0, acc0, false, false);
        acc1 = __builtin_amdgcn_wmma_f32_16x16x32_f16(false, a1.v, false, bf,
                                                      (short)0, acc1, false, false);
    }

    // D layout: lane<16 -> M=r, lane>=16 -> M=8+r; N = lane&15
    const int ncol = col0 + lm;
    const float sc = scale[ncol];
    const float bi = bias[ncol];
#pragma unroll
    for (int r = 0; r < 8; ++r) {
        int m0 = row0 + lh * 8 + r;
        int m1 = m0 + 16;
        float y0 = fmaxf(acc0[r] * sc + bi, 0.f);
        float y1 = fmaxf(acc1[r] * sc + bi, 0.f);
        if (outH) {
            outH[(size_t)m0 * ldc + ncol] = (_Float16)y0;
            outH[(size_t)m1 * ldc + ncol] = (_Float16)y1;
        } else {
            outF[(size_t)m0 * ldc + ncol] = y0;
            outF[(size_t)m1 * ldc + ncol] = y1;
        }
    }
}

// ---------------- final 400->1 dot + sigmoid ----------------
__global__ __launch_bounds__(256) void onn_final(
    const float* __restrict__ h2, const float* __restrict__ wout,
    const float* __restrict__ bout, float* __restrict__ out)
{
    int wid  = blockIdx.x * (blockDim.x >> 5) + (threadIdx.x >> 5);
    int lane = threadIdx.x & 31;
    if (wid >= BATCH) return;
    const float* row = h2 + (size_t)wid * D1;
    float s = 0.f;
    for (int k = lane; k < D1; k += 32) s += row[k] * wout[k];
#pragma unroll
    for (int off = 16; off > 0; off >>= 1) s += __shfl_down(s, off);
    if (lane == 0) out[wid] = 1.f / (1.f + __expf(-(s + bout[0])));
}

// ---------------- launch ----------------
extern "C" void kernel_launch(void* const* d_in, const int* in_sizes, int n_in,
                              void* d_out, int out_size, void* d_ws, size_t ws_size,
                              hipStream_t stream)
{
    const int*   x    = (const int*)  d_in[0];
    const float* emb  = (const float*)d_in[1];
    const float* w0   = (const float*)d_in[2];
    const float* b0   = (const float*)d_in[3];
    const float* g0   = (const float*)d_in[4];
    const float* bb0  = (const float*)d_in[5];
    const float* m0   = (const float*)d_in[6];
    const float* v0   = (const float*)d_in[7];
    const float* w1   = (const float*)d_in[8];
    const float* b1   = (const float*)d_in[9];
    const float* g1   = (const float*)d_in[10];
    const float* bb1  = (const float*)d_in[11];
    const float* m1   = (const float*)d_in[12];
    const float* v1   = (const float*)d_in[13];
    const float* wout = (const float*)d_in[14];
    const float* bout = (const float*)d_in[15];
    float* out = (float*)d_out;

    char* ws = (char*)d_ws;
    _Float16* h0  = (_Float16*)(ws + OFF_H0);
    _Float16* w0t = (_Float16*)(ws + OFF_W0T);
    _Float16* w1t = (_Float16*)(ws + OFF_W1T);
    float*    s0  = (float*)(ws + OFF_S0);
    float*    t0  = (float*)(ws + OFF_T0);
    float*    s1  = (float*)(ws + OFF_S1);
    float*    t1  = (float*)(ws + OFF_T1);
    _Float16* h1  = (_Float16*)(ws + OFF_H1);
    float*    h2  = (float*)(ws + OFF_H2);

    // BN fold (both layers)
    onn_fold_bn<<<(D0 + 255) / 256, 256, 0, stream>>>(b0, g0, bb0, m0, v0, s0, t0, D0);
    onn_fold_bn<<<(D1 + 255) / 256, 256, 0, stream>>>(b1, g1, bb1, m1, v1, s1, t1, D1);

    // Weight transpose/convert
    onn_wt_cvt<<<(D0 * K0PAD + 255) / 256, 256, 0, stream>>>(w0, w0t, IN_DIM, D0, K0PAD);
    onn_wt_cvt<<<(D1 * K1PAD + 255) / 256, 256, 0, stream>>>(w1, w1t, D0, D1, K1PAD);

    // Gather + FFM -> h0
    onn_gather_ffm<<<BATCH, 256, 0, stream>>>(x, emb, h0);

    // Zero h1 so pad columns (400..415) are zero for layer-1 K loop
    hipMemsetAsync(h1, 0, SZ_H1, stream);

    // Layer 0: h0 [4096][640] x w0t -> h1 f16 [4096][416]
    {
        int waves  = (BATCH >> 5) * (D0 >> 4);       // 128 * 25 = 3200
        int blocks = (waves + 7) / 8;                // 8 waves / block
        onn_gemm_bn_relu<<<blocks, 256, 0, stream>>>(h0, w0t, s0, t0,
                                                     h1, nullptr,
                                                     BATCH, D0, K0PAD, K1PAD);
    }

    // Layer 1: h1 [4096][416] x w1t -> h2 f32 [4096][400]
    {
        int waves  = (BATCH >> 5) * (D1 >> 4);
        int blocks = (waves + 7) / 8;
        onn_gemm_bn_relu<<<blocks, 256, 0, stream>>>(h1, w1t, s1, t1,
                                                     nullptr, h2,
                                                     BATCH, D1, K1PAD, D1);
    }

    // Output layer + sigmoid
    onn_final<<<(BATCH + 7) / 8, 256, 0, stream>>>(h2, wout, bout, out);
}
